// GNN_12661563588637
// MI455X (gfx1250) — matrix operand order repaired and checked by reading, not compiled, not run on hardware
//
#include <hip/hip_runtime.h>
#include <hip/hip_bf16.h>

typedef __attribute__((ext_vector_type(16))) _Float16 v16h;
typedef __attribute__((ext_vector_type(8)))  float    v8f;

#define N_NODES 50000
#define N_EDGES 800000
#define HID     64
#define N_CLASS 40
#define N_MASK  5000
#define NUM_PROP 5
#define N_TILES (N_NODES/16)            // 3125 node tiles of 16
#define EPSF 1.1920928955078125e-07f
#define NUMEL ((float)(N_NODES)*(float)(HID))

// ---------------- utility kernels ----------------
__global__ void k_zero(float* __restrict__ p, int n) {
    int i = blockIdx.x * blockDim.x + threadIdx.x;
    if (i < n) p[i] = 0.0f;
}

// pack W_ih / W_hh (192x64 row-major f32) into WMMA B-fragment order, f16:
// Bp[((nt*2+kk)*32 + lane)*16 + e] = W[n*64 + k],  n = nt*16 + lane%16,
// k = kk*32 + 8*(lane/16) + (e%8) + 16*(e/8)   (B 32x16 f16 layout: lanes 0-15 K=0-7,16-23; 16-31 K=8-15,24-31)
__global__ void k_pack(const float* __restrict__ Wih, const float* __restrict__ Whh,
                       _Float16* __restrict__ BpI, _Float16* __restrict__ BpH) {
    int tid = blockIdx.x * blockDim.x + threadIdx.x;         // 0 .. 24575
    if (tid >= 2 * 12288) return;
    int which = tid / 12288;
    int f     = tid % 12288;
    int e    = f & 15;
    int lane = (f >> 4) & 31;
    int kk   = (f >> 9) & 1;
    int nt   = f >> 10;                                      // 0..11
    int n = nt * 16 + (lane & 15);
    int k = kk * 32 + 8 * (lane >> 4) + (e & 7) + 16 * (e >> 3);
    const float* W = which ? Whh : Wih;
    _Float16*    B = which ? BpH : BpI;
    B[f] = (_Float16)W[n * HID + k];
}

__global__ void k_deg(const int* __restrict__ edge, float* __restrict__ deg) {
    int e = blockIdx.x * blockDim.x + threadIdx.x;
    if (e < N_EDGES) atomicAdd(&deg[edge[2 * e + 1]], 1.0f);
}

__global__ void k_invdeg(float* __restrict__ deg) {
    int i = blockIdx.x * blockDim.x + threadIdx.x;
    if (i < N_NODES) deg[i] = 1.0f / (deg[i] + EPSF);
}

// msg[dst] += h[src], thread per (edge, group-of-4-channels)
__global__ void k_scatter(const int* __restrict__ edge, const float* __restrict__ h,
                          float* __restrict__ msg) {
    int tid = blockIdx.x * blockDim.x + threadIdx.x;
    if (tid >= N_EDGES * 16) return;
    int e = tid >> 4;
    int q = tid & 15;
    int src = edge[2 * e];
    int dst = edge[2 * e + 1];
    float4 v = *(const float4*)(h + (size_t)src * HID + q * 4);
    float* mrow = msg + (size_t)dst * HID + q * 4;
    atomicAdd(mrow + 0, v.x);
    atomicAdd(mrow + 1, v.y);
    atomicAdd(mrow + 2, v.z);
    atomicAdd(mrow + 3, v.w);
}

// Build 16x32 f16 A-fragment for this lane from a 32-float chunk of row `m`.
// Layout: m = lane%16, half = lane/16; element e -> k = e + 8*(e>=8) + 8*half,
// i.e. two contiguous runs of 8 floats: [8*half .. +7] and [16+8*half .. +7].
__device__ inline v16h load_afrag(const float* __restrict__ row, int half, float scale) {
    float t[16];
    const float4* p0 = (const float4*)(row + 8 * half);
    const float4* p1 = (const float4*)(row + 16 + 8 * half);
    *(float4*)(t + 0)  = p0[0];
    *(float4*)(t + 4)  = p0[1];
    *(float4*)(t + 8)  = p1[0];
    *(float4*)(t + 12) = p1[1];
    v16h a;
#pragma unroll
    for (int i = 0; i < 16; ++i) a[i] = (_Float16)(t[i] * scale);
    return a;
}

__device__ inline float sigm(float x) { return 1.0f / (1.0f + expf(-x)); }

// One wave per 16-node tile: gi = (msg*invdeg)@W_ih^T, gh = h@W_hh^T via WMMA,
// then GRU gates, h_new store, and squared-diff accumulation.
__global__ void k_gru(const float* __restrict__ msg, const float* __restrict__ hcur,
                      float* __restrict__ hnext,
                      const _Float16* __restrict__ BpI, const _Float16* __restrict__ BpH,
                      const float* __restrict__ invdeg,
                      const float* __restrict__ b_ih, const float* __restrict__ b_hh,
                      float* __restrict__ diffacc) {
    int wave = (blockIdx.x * blockDim.x + threadIdx.x) >> 5;
    int lane = threadIdx.x & 31;
    if (wave >= N_TILES) return;                    // wave-uniform exit, EXEC stays full
    int base = wave * 16;
    int m    = lane & 15;
    int half = lane >> 4;

    float sc = invdeg[base + m];
    const float* mrow = msg  + (size_t)(base + m) * HID;
    const float* hrow = hcur + (size_t)(base + m) * HID;

    v16h aM[2], aH[2];
    aM[0] = load_afrag(mrow,      half, sc);
    aM[1] = load_afrag(mrow + 32, half, sc);
    aH[0] = load_afrag(hrow,      half, 1.0f);
    aH[1] = load_afrag(hrow + 32, half, 1.0f);

    float sq = 0.0f;
#pragma unroll
    for (int ct = 0; ct < 4; ++ct) {                // 4 column tiles of 16 channels
        v8f accI[3], accH[3];
#pragma unroll
        for (int g = 0; g < 3; ++g) {               // r, z, n gate blocks
            v8f ai = {}; v8f ah = {};
            int nt = g * 4 + ct;                    // column tile in 0..11
#pragma unroll
            for (int kk = 0; kk < 2; ++kk) {        // K = 64 = 2 x 32
                v16h bi = *(const v16h*)(BpI + ((size_t)(nt * 2 + kk) * 32 + lane) * 16);
                v16h bh = *(const v16h*)(BpH + ((size_t)(nt * 2 + kk) * 32 + lane) * 16);
                ai = __builtin_amdgcn_wmma_f32_16x16x32_f16(false, aM[kk], false, bi,
                                                            (short)0, ai, false, false);
                ah = __builtin_amdgcn_wmma_f32_16x16x32_f16(false, aH[kk], false, bh,
                                                            (short)0, ah, false, false);
            }
            accI[g] = ai;
            accH[g] = ah;
        }
        int c = ct * 16 + (lane & 15);
        float bir = b_ih[c],       bhr = b_hh[c];
        float biz = b_ih[64 + c],  bhz = b_hh[64 + c];
        float bin = b_ih[128 + c], bhn = b_hh[128 + c];
#pragma unroll
        for (int v = 0; v < 8; ++v) {               // D layout: M = v + 8*half, N = lane%16
            int node = base + v + 8 * half;
            float i_r = accI[0][v] + bir, h_r = accH[0][v] + bhr;
            float i_z = accI[1][v] + biz, h_z = accH[1][v] + bhz;
            float i_n = accI[2][v] + bin, h_n = accH[2][v] + bhn;
            float r = sigm(i_r + h_r);
            float z = sigm(i_z + h_z);
            float n = tanhf(i_n + r * h_n);
            float hold = hcur[(size_t)node * HID + c];
            float hnew = (1.0f - z) * n + z * hold;
            hnext[(size_t)node * HID + c] = hnew;
            float d = hnew - hold;
            sq += d * d;
        }
    }
#pragma unroll
    for (int off = 16; off > 0; off >>= 1) sq += __shfl_xor(sq, off, 32);
    if (lane == 0) atomicAdd(diffacc, sq);
}

// y_m[i][n] = h[mask[i]] . W_out[n] + b_out[n]
__global__ void k_out(const float* __restrict__ h, const float* __restrict__ Wout,
                      const float* __restrict__ bout, const int* __restrict__ mask,
                      float* __restrict__ y) {
    int tid = blockIdx.x * blockDim.x + threadIdx.x;
    if (tid >= N_MASK * N_CLASS) return;
    int i = tid / N_CLASS;
    int n = tid % N_CLASS;
    const float* hr = h + (size_t)mask[i] * HID;
    const float* wr = Wout + (size_t)n * HID;
    float s = bout[n];
#pragma unroll
    for (int k = 0; k < HID; k += 4) {
        float4 a = *(const float4*)(hr + k);
        float4 w = *(const float4*)(wr + k);
        s += a.x * w.x + a.y * w.y + a.z * w.z + a.w * w.w;
    }
    y[tid] = s;
}

__global__ void k_loss(const float* __restrict__ y, const int* __restrict__ target,
                       const int* __restrict__ mask, float* __restrict__ tm_out,
                       float* __restrict__ lossacc) {
    int i = blockIdx.x * blockDim.x + threadIdx.x;
    if (i >= N_MASK) return;
    const float* r = y + (size_t)i * N_CLASS;
    float mx = r[0];
    for (int n = 1; n < N_CLASS; ++n) mx = fmaxf(mx, r[n]);
    float se = 0.0f;
    for (int n = 0; n < N_CLASS; ++n) se += expf(r[n] - mx);
    int t = target[mask[i]];
    float lp = r[t] - mx - logf(se);
    tm_out[i] = (float)t;
    atomicAdd(lossacc, -lp * (1.0f / (float)N_MASK));
}

__global__ void k_fin(const float* __restrict__ diffacc, const float* __restrict__ lossacc,
                      float* __restrict__ out) {
    int i = threadIdx.x;
    if (i < NUM_PROP)
        out[N_MASK * N_CLASS + N_MASK + i] = sqrtf(diffacc[i]) * (1.0f / NUMEL);
    if (i == NUM_PROP)
        out[N_MASK * N_CLASS + N_MASK + NUM_PROP] = lossacc[0];
}

// ---------------- host launcher ----------------
extern "C" void kernel_launch(void* const* d_in, const int* in_sizes, int n_in,
                              void* d_out, int out_size, void* d_ws, size_t ws_size,
                              hipStream_t stream) {
    (void)in_sizes; (void)n_in; (void)out_size; (void)ws_size;
    const float* feat  = (const float*)d_in[0];
    const float* W_ih  = (const float*)d_in[1];
    const float* W_hh  = (const float*)d_in[2];
    const float* b_ih  = (const float*)d_in[3];
    const float* b_hh  = (const float*)d_in[4];
    const float* W_out = (const float*)d_in[5];
    const float* b_out = (const float*)d_in[6];
    const int*   edge  = (const int*)d_in[7];
    const int*   target= (const int*)d_in[8];
    const int*   mask  = (const int*)d_in[9];
    float* out = (float*)d_out;

    const size_t HN = (size_t)N_NODES * HID;        // 3.2M floats
    float* ws      = (float*)d_ws;
    float* hA      = ws;
    float* hB      = hA + HN;
    float* msg     = hB + HN;
    float* invdeg  = msg + HN;
    float* diffacc = invdeg + N_NODES;              // 8 slots
    float* lossacc = diffacc + 8;                   // 8 slots
    _Float16* BpI  = (_Float16*)(lossacc + 8);      // 12288 halves, 32B-aligned
    _Float16* BpH  = BpI + 12288;

    // init: deg accumulator, diff/loss accumulators; h0 = feat; pack weights
    k_zero<<<(N_NODES + 255) / 256, 256, 0, stream>>>(invdeg, N_NODES);
    k_zero<<<1, 64, 0, stream>>>(diffacc, 16);
    hipMemcpyAsync(hA, feat, HN * sizeof(float), hipMemcpyDeviceToDevice, stream);
    k_pack<<<(2 * 12288 + 255) / 256, 256, 0, stream>>>(W_ih, W_hh, BpI, BpH);
    k_deg<<<(N_EDGES + 255) / 256, 256, 0, stream>>>(edge, invdeg);
    k_invdeg<<<(N_NODES + 255) / 256, 256, 0, stream>>>(invdeg);

    float* hcur = hA;
    float* hnext = hB;
    for (int s = 0; s < NUM_PROP; ++s) {
        k_zero<<<((int)HN + 255) / 256, 256, 0, stream>>>(msg, (int)HN);
        k_scatter<<<(N_EDGES * 16 + 255) / 256, 256, 0, stream>>>(edge, hcur, msg);
        k_gru<<<(N_TILES * 32 + 255) / 256, 256, 0, stream>>>(
            msg, hcur, hnext, BpI, BpH, invdeg, b_ih, b_hh, diffacc + s);
        float* t = hcur; hcur = hnext; hnext = t;
    }
    k_out<<<(N_MASK * N_CLASS + 255) / 256, 256, 0, stream>>>(hcur, W_out, b_out, mask, out);
    k_loss<<<(N_MASK + 255) / 256, 256, 0, stream>>>(out, target, mask,
                                                     out + N_MASK * N_CLASS, lossacc);
    k_fin<<<1, 32, 0, stream>>>(diffacc, lossacc, out);
}
